// NeuralMeshFlow_64218351010355
// MI455X (gfx1250) — compile-verified, matrix operand order checked by reading, not run on hardware
//
#include <hip/hip_runtime.h>
#include <hip/hip_bf16.h>

typedef unsigned short u16;
typedef unsigned int   u32;
typedef __attribute__((ext_vector_type(16))) __bf16 v16bf;
typedef __attribute__((ext_vector_type(8)))  float  v8f;

#define B_    4
#define N_    2562
#define H_    512
#define ROWS_ (B_ * N_)          // 10248 rows of (rows,3)
#define TOT_  (ROWS_ * 3)        // 30744 floats in x
#define MT_   64                 // rows per workgroup
#define ROWP_ 520                // padded LDS row stride in halves (1040B, 16B-aligned, bank-staggered)
#define LDSB_ (MT_ * ROWP_)      // halves per buffer
#define KT_   16                 // 512 / 32 k-tiles
#define NBLK_ 6

__device__ __forceinline__ float bf2f(u16 u) {
    u32 x = ((u32)u) << 16; float f; __builtin_memcpy(&f, &x, 4); return f;
}
__device__ __forceinline__ u16 f2bf(float f) {
    u32 x; __builtin_memcpy(&x, &f, 4);
    u32 r = x + 0x7FFFu + ((x >> 16) & 1u);
    return (u16)(r >> 16);
}

union Frag16 { v16bf v; float4 f[2]; };

// ---------------------------------------------------------------------------
// Pack one weight tensor (NBLK, 512, 512) fp32 row-major [K][N] into the exact
// per-lane WMMA B-fragment layout for v_wmma_f32_16x16x32_bf16:
//   dst[blk][tn(32)][tk(16)][lane(32)][e(16)] (bf16)
//   K(e,lane) = e + (e>=8 ? 8:0) + (lane>=16 ? 8:0);  N = tn*16 + (lane&15)
// Each lane of the GEMM then loads its 32 fragment bytes fully coalesced.
// ---------------------------------------------------------------------------
__global__ void nmf_pack_w(const float* __restrict__ W, u16* __restrict__ Wp) {
    int i = blockIdx.x * 256 + threadIdx.x;
    if (i >= (NBLK_ << 18)) return;
    int e    = i & 15;
    int lane = (i >> 4) & 31;
    int tk   = (i >> 9) & 15;
    int tn   = (i >> 13) & 31;
    int blk  = i >> 18;
    int K = tk * 32 + e + ((e >= 8) ? 8 : 0) + ((lane >= 16) ? 8 : 0);
    int n = tn * 16 + (lane & 15);
    Wp[i] = f2bf(W[((size_t)blk * H_ + K) * H_ + n]);
}

// cf[k][b][c] = tanh(clv[b] . Wc[k][:,c] + bc[k][c])
__global__ void nmf_cf(const float* __restrict__ clv, const float* __restrict__ Wc,
                       const float* __restrict__ bc, float* __restrict__ cf) {
    int k = blockIdx.x / B_, b = blockIdx.x % B_;
    for (int c = threadIdx.x; c < H_; c += 256) {
        float s = bc[k * H_ + c];
        for (int K = 0; K < H_; ++K)
            s = fmaf(clv[b * H_ + K], Wc[((size_t)k * H_ + K) * H_ + c], s);
        cf[(k * B_ + b) * H_ + c] = tanhf(s);
    }
}

// scale[j][b][c] = sigmoid(relu(clv[b] @ Wn1[j] + bn1[j]) @ Wn2[j] + bn2[j])
__global__ void nmf_scale(const float* __restrict__ clv, const float* __restrict__ Wn1,
                          const float* __restrict__ bn1, const float* __restrict__ Wn2,
                          const float* __restrict__ bn2, float* __restrict__ scale) {
    int j = blockIdx.x >> 2, b = blockIdx.x & 3;
    __shared__ float h[256];
    int t = threadIdx.x;
    float s = bn1[j * 256 + t];
    for (int K = 0; K < H_; ++K)
        s = fmaf(clv[b * H_ + K], Wn1[((size_t)j * H_ + K) * 256 + t], s);
    h[t] = fmaxf(s, 0.f);
    __syncthreads();
    if (t < 3) {
        float o = bn2[j * 3 + t];
        for (int q = 0; q < 256; ++q)
            o = fmaf(h[q], Wn2[(j * 256 + q) * 3 + t], o);
        scale[(j * B_ + b) * 3 + t] = 1.0f / (1.0f + expf(-o));
    }
}

__global__ void nmf_broadcast(const float* __restrict__ verts, float* __restrict__ x) {
    int i = blockIdx.x * 256 + threadIdx.x;
    if (i >= TOT_) return;
    x[i] = verts[i % (N_ * 3)];
}

// mean over axis=1 for each (b, dim): 12 blocks
__global__ void nmf_mean(const float* __restrict__ x, float* __restrict__ mean) {
    int b = blockIdx.x / 3, d = blockIdx.x % 3;
    __shared__ float red[256];
    float s = 0.f;
    for (int n = threadIdx.x; n < N_; n += 256)
        s += x[(b * N_ + n) * 3 + d];
    red[threadIdx.x] = s;
    __syncthreads();
    for (int st = 128; st > 0; st >>= 1) {
        if (threadIdx.x < st) red[threadIdx.x] += red[threadIdx.x + st];
        __syncthreads();
    }
    if (threadIdx.x == 0) mean[blockIdx.x] = red[0] / (float)N_;
}

__global__ void nmf_adain(float* __restrict__ x, const float* __restrict__ ap,
                          const float* __restrict__ mean, const float* __restrict__ scale,
                          int j, float* __restrict__ outSlot) {
    int i = blockIdx.x * 256 + threadIdx.x;
    if (i >= TOT_) return;
    int d = i % 3;
    int b = (i / 3) / N_;
    float a  = ap[b * 24 + 6 * j + d];
    float bb = ap[b * 24 + 6 * j + 3 + d];
    float sc = scale[(j * B_ + b) * 3 + d];
    float v = a + bb * (x[i] - mean[b * 3 + d]) * (1.0f - sc);
    x[i] = v;
    if (outSlot) outSlot[i] = v;
}

__global__ void nmf_rk(float* __restrict__ x, const float* __restrict__ k1,
                       const float* __restrict__ k2, const float* __restrict__ k3,
                       const float* __restrict__ k4, float dt6) {
    int i = blockIdx.x * 256 + threadIdx.x;
    if (i >= TOT_) return;
    x[i] += dt6 * (k1[i] + 2.f * (k2[i] + k3[i]) + k4[i]);
}

// ---------------------------------------------------------------------------
// One 512x512 bf16 WMMA layer with fused bias+relu+residual, LDS->LDS.
// Wave w: m-tile = w&3 (16 rows), n-half = w>>2 (256 cols = 16 n-tiles).
// ---------------------------------------------------------------------------
__device__ __forceinline__ void nmf_gemm_layer(const u16* __restrict__ in, u16* __restrict__ out,
                                               const u16* __restrict__ Wp,
                                               const float* __restrict__ bias,
                                               int lane, int w) {
    const int mt = w & 3;
    const int nHalf = w >> 2;
    const v8f zero = {0.f, 0.f, 0.f, 0.f, 0.f, 0.f, 0.f, 0.f};
    v8f acc[16];
#pragma unroll
    for (int i = 0; i < 16; ++i) acc[i] = zero;

    const int arow = mt * 16 + (lane & 15);
    const int ksub = (lane >= 16) ? 8 : 0;

    for (int kt = 0; kt < KT_; ++kt) {
        Frag16 a;
        const int off = arow * ROWP_ + kt * 32 + ksub;
        a.f[0] = *reinterpret_cast<const float4*>(in + off);
        a.f[1] = *reinterpret_cast<const float4*>(in + off + 16);
#pragma unroll
        for (int nt = 0; nt < 16; ++nt) {
            const int tn = nHalf * 16 + nt;
            const float4* bp =
                reinterpret_cast<const float4*>(Wp + (((size_t)tn * KT_ + kt) * 32 + lane) * 16);
            Frag16 bfr;
            bfr.f[0] = bp[0];
            bfr.f[1] = bp[1];
            acc[nt] = __builtin_amdgcn_wmma_f32_16x16x32_bf16(
                false, a.v, false, bfr.v, (short)0, acc[nt], false, false);
        }
    }

    // Epilogue: D layout — lane L holds N = L%16; VGPR v holds M = v + (L>=16 ? 8:0).
    const int nloc = lane & 15;
    const int mofs = (lane >= 16) ? 8 : 0;
#pragma unroll
    for (int nt = 0; nt < 16; ++nt) {
        const int n = (nHalf * 16 + nt) * 16 + nloc;
        const float bn = bias[n];
#pragma unroll
        for (int v = 0; v < 8; ++v) {
            const int m = mt * 16 + v + mofs;
            float val = fmaxf(acc[nt][v] + bn, 0.f) + bf2f(in[m * ROWP_ + n]);
            out[m * ROWP_ + n] = f2bf(val);
        }
    }
}

// ---------------------------------------------------------------------------
// Fused _dyn eval: kout = tanh(W4 . (resid chain of relu(W1..)) + b4)
// input to f is x + coef*kprev (coef=0 and kprev=x for stage k1).
// ---------------------------------------------------------------------------
__global__ void __launch_bounds__(256)
nmf_dyn(const float* __restrict__ x, const float* __restrict__ kprev, float coef,
        const float* __restrict__ cf,                  // [B][512] for this block k
        const float* __restrict__ W1, const float* __restrict__ b1,
        const u16* __restrict__ W2p, const float* __restrict__ b2,
        const u16* __restrict__ W3p, const float* __restrict__ b3,
        const float* __restrict__ W4, const float* __restrict__ b4,
        float* __restrict__ kout) {
    extern __shared__ u16 lds[];
    u16* bufA = lds;
    u16* bufB = lds + LDSB_;

    const int tid = threadIdx.x;
    const int lane = tid & 31;
    const int w = tid >> 5;
    const int rowBase = blockIdx.x * MT_;

    // Phase 1: h0 = relu((x + coef*kprev) @ W1 + b1) * cf  -> bufA (bf16)
    for (int i = tid; i < MT_ * H_; i += 256) {
        const int r = i >> 9;
        const int c = i & (H_ - 1);
        const int gr = rowBase + r;
        float v = 0.f;
        if (gr < ROWS_) {
            const int b = gr / N_;
            const float x0 = x[gr * 3 + 0] + coef * kprev[gr * 3 + 0];
            const float x1 = x[gr * 3 + 1] + coef * kprev[gr * 3 + 1];
            const float x2 = x[gr * 3 + 2] + coef * kprev[gr * 3 + 2];
            float h = b1[c];
            h = fmaf(x0, W1[c], h);
            h = fmaf(x1, W1[H_ + c], h);
            h = fmaf(x2, W1[2 * H_ + c], h);
            v = fmaxf(h, 0.f) * cf[b * H_ + c];
        }
        bufA[r * ROWP_ + c] = f2bf(v);
    }
    __syncthreads();

    // Phase 2: h1 = relu(h0 @ W2 + b2) + h0   (bufA -> bufB)
    nmf_gemm_layer(bufA, bufB, W2p, b2, lane, w);
    __syncthreads();

    // Phase 3: h2 = relu(h1 @ W3 + b3) + h1   (bufB -> bufA)
    nmf_gemm_layer(bufB, bufA, W3p, b3, lane, w);
    __syncthreads();

    // Phase 4: kout = tanh(h2 @ W4 + b4)
    if (tid < MT_ * 3) {
        const int r = tid / 3, c = tid % 3;
        const int gr = rowBase + r;
        if (gr < ROWS_) {
            float s = b4[c];
            for (int K = 0; K < H_; ++K)
                s = fmaf(bf2f(bufA[r * ROWP_ + K]), W4[K * 3 + c], s);
            kout[gr * 3 + c] = tanhf(s);
        }
    }
}

// ---------------------------------------------------------------------------
extern "C" void kernel_launch(void* const* d_in, const int* in_sizes, int n_in,
                              void* d_out, int out_size, void* d_ws, size_t ws_size,
                              hipStream_t stream) {
    const float* clv  = (const float*)d_in[0];
    const float* ap   = (const float*)d_in[1];
    const float* verts= (const float*)d_in[2];
    const float* W1   = (const float*)d_in[3];
    const float* b1   = (const float*)d_in[4];
    const float* W2   = (const float*)d_in[5];
    const float* b2   = (const float*)d_in[6];
    const float* W3   = (const float*)d_in[7];
    const float* b3   = (const float*)d_in[8];
    const float* W4   = (const float*)d_in[9];
    const float* b4   = (const float*)d_in[10];
    const float* Wc   = (const float*)d_in[11];
    const float* bc   = (const float*)d_in[12];
    const float* Wn1  = (const float*)d_in[13];
    const float* bn1  = (const float*)d_in[14];
    const float* Wn2  = (const float*)d_in[15];
    const float* bn2  = (const float*)d_in[16];
    float* out = (float*)d_out;

    char* ws = (char*)d_ws;
    auto alloc = [&](size_t bytes) -> char* {
        char* p = ws;
        ws += (bytes + 255) & ~(size_t)255;
        return p;
    };
    float* xb  = (float*)alloc(TOT_ * 4);
    float* k1  = (float*)alloc(TOT_ * 4);
    float* k2  = (float*)alloc(TOT_ * 4);
    float* k3  = (float*)alloc(TOT_ * 4);
    float* k4  = (float*)alloc(TOT_ * 4);
    float* cfb = (float*)alloc((size_t)NBLK_ * B_ * H_ * 4);
    float* scb = (float*)alloc(4 * B_ * 3 * 4);
    float* mnb = (float*)alloc(B_ * 3 * 4);
    u16* W2p   = (u16*)alloc((size_t)NBLK_ * 262144 * 2);
    u16* W3p   = (u16*)alloc((size_t)NBLK_ * 262144 * 2);

    const int packElems = NBLK_ << 18;
    nmf_pack_w<<<(packElems + 255) / 256, 256, 0, stream>>>(W2, W2p);
    nmf_pack_w<<<(packElems + 255) / 256, 256, 0, stream>>>(W3, W3p);
    nmf_cf<<<NBLK_ * B_, 256, 0, stream>>>(clv, Wc, bc, cfb);
    nmf_scale<<<4 * B_, 256, 0, stream>>>(clv, Wn1, bn1, Wn2, bn2, scb);

    const int eb = (TOT_ + 255) / 256;
    nmf_broadcast<<<eb, 256, 0, stream>>>(verts, xb);
    nmf_mean<<<B_ * 3, 256, 0, stream>>>(xb, mnb);
    nmf_adain<<<eb, 256, 0, stream>>>(xb, ap, mnb, scb, 0, nullptr);

    const float dt = 0.2f / 4.0f;
    const int nWG = (ROWS_ + MT_ - 1) / MT_;
    const size_t shmem = (size_t)2 * LDSB_ * 2;

    for (int db = 0; db < 3; ++db) {
        for (int kk = 2 * db; kk <= 2 * db + 1; ++kk) {
            const float* cfk = cfb + (size_t)kk * B_ * H_;
            const float* W1k = W1 + (size_t)kk * 3 * H_;
            const float* b1k = b1 + (size_t)kk * H_;
            const u16* W2pk  = W2p + ((size_t)kk << 18);
            const float* b2k = b2 + (size_t)kk * H_;
            const u16* W3pk  = W3p + ((size_t)kk << 18);
            const float* b3k = b3 + (size_t)kk * H_;
            const float* W4k = W4 + (size_t)kk * H_ * 3;
            const float* b4k = b4 + (size_t)kk * 3;
            for (int st = 0; st < 4; ++st) {
                nmf_dyn<<<nWG, 256, shmem, stream>>>(xb, xb, 0.0f, cfk, W1k, b1k,
                                                     W2pk, b2k, W3pk, b3k, W4k, b4k, k1);
                nmf_dyn<<<nWG, 256, shmem, stream>>>(xb, k1, 0.5f * dt, cfk, W1k, b1k,
                                                     W2pk, b2k, W3pk, b3k, W4k, b4k, k2);
                nmf_dyn<<<nWG, 256, shmem, stream>>>(xb, k2, 0.5f * dt, cfk, W1k, b1k,
                                                     W2pk, b2k, W3pk, b3k, W4k, b4k, k3);
                nmf_dyn<<<nWG, 256, shmem, stream>>>(xb, k3, dt, cfk, W1k, b1k,
                                                     W2pk, b2k, W3pk, b3k, W4k, b4k, k4);
                nmf_rk<<<eb, 256, 0, stream>>>(xb, k1, k2, k3, k4, dt / 6.0f);
            }
        }
        const int j = db + 1;
        nmf_mean<<<B_ * 3, 256, 0, stream>>>(xb, mnb);
        nmf_adain<<<eb, 256, 0, stream>>>(xb, ap, mnb, scb, j, out + (size_t)db * TOT_);
    }
}
